// SparseAttention_83373905150280
// MI455X (gfx1250) — matrix operand order repaired and checked by reading, not compiled, not run on hardware
//
#include <hip/hip_runtime.h>
#include <hip/hip_bf16.h>

// Problem constants (fixed by the reference)
#define BB     4
#define LL     4096
#define DD     512
#define HH     8
#define DHD    64
#define KNB    16
#define ROWS   (BB * LL)      // 16384

// GEMM tiling: 256 threads = 8 waves; each wave owns a 16x128 slab of C
// (8 x v_wmma_f32_16x16x32_bf16 per K-step from one A fragment).
#define BM 128
#define BN 128
#define BK 32
#define LDAU 20               // u32 per As row: 16 data + 4 pad (80B, 16B-aligned, conflict-free)
#define LDBU 20               // u32 per Bs row (n-major)
#define NKT (DD / BK)         // 16 K-steps

typedef unsigned int u32;
typedef __attribute__((ext_vector_type(16))) __bf16 v16bf;
typedef __attribute__((ext_vector_type(8)))  float  v8f;

// Pack two f32 into one u32 of bf16 pairs (lowers to v_cvt_pk_bf16_f32).
__device__ __forceinline__ u32 pkbf(float x, float y) {
  union { __bf16 b[2]; u32 u; } t;
  t.b[0] = (__bf16)x; t.b[1] = (__bf16)y;
  return t.u;
}

// Two 16B LDS reads -> one 16-element bf16 fragment.
__device__ __forceinline__ v16bf frag_ld(const u32* p0, const u32* p1) {
  union { uint4 q[2]; v16bf v; } t;
  t.q[0] = *(const uint4*)p0;
  t.q[1] = *(const uint4*)p1;
  return t.v;
}

// ---------------------------------------------------------------------------
// Kernel 1: top-16 neighbors per row. One wave32 per query token.
// Per-lane sorted (ascending) top-16 list in LDS (dynamic indexing), padded to
// stride 17 (68B -> 17 banks, conflict-free), then a 32-way lexicographic
// (dist^2, idx) merge via shuffle butterflies. dist^2 preserves dist ordering;
// index tie-break matches jax.lax.top_k stability.
// ---------------------------------------------------------------------------
__global__ __launch_bounds__(256) void topk_kernel(const float* __restrict__ pos,
                                                   int* __restrict__ nbr,
                                                   int* __restrict__ is_in) {
  __shared__ float sd[8][32][KNB + 1];
  __shared__ int   si[8][32][KNB + 1];
  const int w    = threadIdx.x >> 5;
  const int lane = threadIdx.x & 31;
  const int g    = blockIdx.x * 8 + w;       // global row
  const int b    = g >> 12;
  const int l    = g & (LL - 1);
  const float* pb = pos + (size_t)b * LL * 3;
  const float qx = pb[l * 3 + 0], qy = pb[l * 3 + 1], qz = pb[l * 3 + 2];

  float* ld = sd[w][lane];
  int*   li = si[w][lane];
#pragma unroll
  for (int i = 0; i < KNB; ++i) { ld[i] = 3.0e38f; li[i] = 0x7fffffff; }

#pragma unroll 1
  for (int j = lane; j < LL; j += 32) {
    const float dx = pb[j * 3 + 0] - qx;
    const float dy = pb[j * 3 + 1] - qy;
    const float dz = pb[j * 3 + 2] - qz;
    const float d2 = dx * dx + dy * dy + dz * dz;
    if (d2 < ld[KNB - 1]) {
      int p = KNB - 1;
      while (p > 0 && (ld[p - 1] > d2 || (ld[p - 1] == d2 && li[p - 1] > j))) {
        ld[p] = ld[p - 1]; li[p] = li[p - 1]; --p;
      }
      ld[p] = d2; li[p] = j;
    }
  }

  int head = 0;
  int self_in = 0;
#pragma unroll 1
  for (int r = 0; r < KNB; ++r) {
    float d = (head < KNB) ? ld[head] : 3.0e38f;
    int   i = (head < KNB) ? li[head] : 0x7fffffff;
#pragma unroll
    for (int off = 16; off > 0; off >>= 1) {
      const float od = __shfl_xor(d, off, 32);
      const int   oi = __shfl_xor(i, off, 32);
      if (od < d || (od == d && oi < i)) { d = od; i = oi; }
    }
    if (head < KNB && li[head] == i) ++head;   // unique winner advances
    if (lane == r) nbr[(size_t)g * KNB + r] = i;
    if (i == l) self_in = 1;
  }
  if (lane == 0) is_in[g] = self_in;
}

// ---------------------------------------------------------------------------
// Shared GEMM machinery: C[m,n] = sum_k A[m,k] * W[n,k]  (A,W f32 row-major),
// computed as bf16 WMMA with f32 accumulate. Double-buffered LDS, one barrier
// per K-step, all fragment traffic as ds_load_b128 / ds_store_b128.
//
// WMMA layouts (ISA 7.12.2, wave32, 16-bit):
//   A 16x32: lane m=(l&15), hf=l>>4; u32 word v(0..7) holds K pair; word index
//            in a K-major row: {hf*4+v} for v<4, {8+hf*4+(v-4)} for v>=4.
//   B 32x16: lane n=(l&15), hf=l>>4; u32 word v holds K pair; word index
//            in an n-major row: {hf*4+v} for v<4, {8+hf*4+(v-4)} for v>=4.
// ---------------------------------------------------------------------------
struct GemmSlots {
  u32 (*As)[BM][LDAU];   // [2][BM][LDAU]
  u32 (*Bs)[BN][LDBU];   // [2][BN][LDBU]
};

__device__ __forceinline__ void gemm_core(const float* __restrict__ A,
                                          const float* __restrict__ W,
                                          int m0, int n0, GemmSlots s,
                                          v8f acc[8]) {
  const int tid  = threadIdx.x;
  const int wave = tid >> 5, lane = tid & 31;

  // A loader: 2 threads per row, 16 consecutive f32 each.
  const int arow  = tid >> 1;
  const int acol  = (tid & 1) * 16;
  const int acol2 = acol >> 1;
  // B loader: 2 threads per W row, 16 consecutive f32 each.
  const int brow  = tid >> 1;
  const int bcol  = (tid & 1) * 16;
  const int bcol2 = bcol >> 1;

  const float* asrc = A + (size_t)(m0 + arow) * DD + acol;
  const float* bsrc = W + (size_t)(n0 + brow) * DD + bcol;

  const int am  = wave * 16 + (lane & 15);
  const int hf  = lane >> 4;
  const int hf4 = hf * 4;
  const int hf8 = hf * 8;

  float4 ar[4], br[4];
#pragma unroll
  for (int i = 0; i < 4; ++i) {
    ar[i] = *(const float4*)(asrc + 4 * i);
    br[i] = *(const float4*)(bsrc + 4 * i);
  }

  // Convert + store one staged tile into LDS buffer `buf`.
  auto stage = [&](int buf, const float4* a4, const float4* b4) {
    uint4 w0 = make_uint4(pkbf(a4[0].x, a4[0].y), pkbf(a4[0].z, a4[0].w),
                          pkbf(a4[1].x, a4[1].y), pkbf(a4[1].z, a4[1].w));
    uint4 w1 = make_uint4(pkbf(a4[2].x, a4[2].y), pkbf(a4[2].z, a4[2].w),
                          pkbf(a4[3].x, a4[3].y), pkbf(a4[3].z, a4[3].w));
    *(uint4*)&s.As[buf][arow][acol2 + 0] = w0;
    *(uint4*)&s.As[buf][arow][acol2 + 4] = w1;
    uint4 u0 = make_uint4(pkbf(b4[0].x, b4[0].y), pkbf(b4[0].z, b4[0].w),
                          pkbf(b4[1].x, b4[1].y), pkbf(b4[1].z, b4[1].w));
    uint4 u1 = make_uint4(pkbf(b4[2].x, b4[2].y), pkbf(b4[2].z, b4[2].w),
                          pkbf(b4[3].x, b4[3].y), pkbf(b4[3].z, b4[3].w));
    *(uint4*)&s.Bs[buf][brow][bcol2 + 0] = u0;
    *(uint4*)&s.Bs[buf][brow][bcol2 + 4] = u1;
  };

  stage(0, ar, br);
  __syncthreads();

#pragma unroll 1
  for (int kt = 0; kt < NKT; ++kt) {
    const int cur = kt & 1;
    const bool more = (kt + 1 < NKT);
    if (more) {
      const int ko = (kt + 1) * BK;
#pragma unroll
      for (int i = 0; i < 4; ++i) {
        ar[i] = *(const float4*)(asrc + ko + 4 * i);
        br[i] = *(const float4*)(bsrc + ko + 4 * i);
      }
    }

    const v16bf a = frag_ld(&s.As[cur][am][hf4], &s.As[cur][am][8 + hf4]);
#pragma unroll
    for (int nt = 0; nt < 8; ++nt) {
      const int bn = nt * 16 + (lane & 15);
      const v16bf bf = frag_ld(&s.Bs[cur][bn][hf4], &s.Bs[cur][bn][8 + hf4]);
      acc[nt] = __builtin_amdgcn_wmma_f32_16x16x32_bf16(false, a, false, bf,
                                                        (short)0, acc[nt], false, false);
    }

    if (more) stage(cur ^ 1, ar, br);
    __syncthreads();
  }
}

// ---------------------------------------------------------------------------
// Kernel 2: fused QKV projection, N = 1536 (Q|K|V); n0 is 128-aligned so each
// block falls entirely in one segment. Epilogue adds bias and the 1/sqrt(dh)
// scale on the Q segment.
// ---------------------------------------------------------------------------
__global__ __launch_bounds__(256) void qkv_gemm_kernel(const float* __restrict__ X,
                                                       const float* __restrict__ W,
                                                       const float* __restrict__ bias,
                                                       float* __restrict__ Q,
                                                       float* __restrict__ Kp,
                                                       float* __restrict__ Vp) {
  __shared__ u32 As[2][BM][LDAU];
  __shared__ u32 Bs[2][BN][LDBU];
  const int wave = threadIdx.x >> 5, lane = threadIdx.x & 31;
  const int m0 = blockIdx.y * BM;
  const int n0 = blockIdx.x * BN;

  v8f acc[8] = {};
  gemm_core(X, W, m0, n0, GemmSlots{As, Bs}, acc);

  const int seg = n0 >> 9;                       // 0=Q, 1=K, 2=V
  float* outb = (seg == 0) ? Q : (seg == 1 ? Kp : Vp);
  const float sc = (seg == 0) ? 0.125f : 1.0f;   // 1/sqrt(64)
#pragma unroll
  for (int nt = 0; nt < 8; ++nt) {
    const int nglob = n0 + nt * 16 + (lane & 15);
    const int n     = nglob & (DD - 1);
    const float bv  = bias[nglob];
#pragma unroll
    for (int r = 0; r < 8; ++r) {
      const int m = m0 + wave * 16 + ((lane >> 4) << 3) + r;
      outb[(size_t)m * DD + n] = (acc[nt][r] + bv) * sc;
    }
  }
}

// ---------------------------------------------------------------------------
// Kernel 3: sparse attention core. One wave32 per token row; each lane owns
// 2 of the 64 head dims. Scores via shuffle-reduced dots; softmax; ctx.
// ---------------------------------------------------------------------------
__global__ __launch_bounds__(256) void attn_kernel(const float* __restrict__ Q,
                                                   const float* __restrict__ Kp,
                                                   const float* __restrict__ Vp,
                                                   const int* __restrict__ nbr,
                                                   float* __restrict__ ctx) {
  const int wave = threadIdx.x >> 5, lane = threadIdx.x & 31;
  const int g = blockIdx.x * 8 + wave;
  const int b = g >> 12;
  const size_t rowOff = (size_t)g * DD;

  int nidx[KNB];
#pragma unroll
  for (int j = 0; j < KNB; ++j)
    nidx[j] = (b << 12) + nbr[(size_t)g * KNB + j];

  const int d0 = lane * 2;
#pragma unroll 1
  for (int h = 0; h < HH; ++h) {
    const int c0 = h * DHD + d0;
    const float q0 = Q[rowOff + c0];
    const float q1 = Q[rowOff + c0 + 1];

    float s[KNB];
#pragma unroll
    for (int j = 0; j < KNB; ++j) {
      const float* kr = Kp + (size_t)nidx[j] * DD + c0;
      float p = q0 * kr[0] + q1 * kr[1];
#pragma unroll
      for (int off = 16; off > 0; off >>= 1) p += __shfl_xor(p, off, 32);
      s[j] = p;
    }
    float mx = s[0];
#pragma unroll
    for (int j = 1; j < KNB; ++j) mx = fmaxf(mx, s[j]);
    float sum = 0.f;
#pragma unroll
    for (int j = 0; j < KNB; ++j) { s[j] = __expf(s[j] - mx); sum += s[j]; }
    const float inv = 1.0f / sum;

    float c0v = 0.f, c1v = 0.f;
#pragma unroll
    for (int j = 0; j < KNB; ++j) {
      const float* vr = Vp + (size_t)nidx[j] * DD + c0;
      const float p = s[j] * inv;
      c0v += p * vr[0];
      c1v += p * vr[1];
    }
    ctx[rowOff + c0]     = c0v;
    ctx[rowOff + c0 + 1] = c1v;
  }
}

// ---------------------------------------------------------------------------
// Kernel 4: output projection + where(is_in, out, x) mask.
// ---------------------------------------------------------------------------
__global__ __launch_bounds__(256) void out_gemm_kernel(const float* __restrict__ Cx,
                                                       const float* __restrict__ Wo,
                                                       const float* __restrict__ bo,
                                                       const float* __restrict__ X,
                                                       const int* __restrict__ is_in,
                                                       float* __restrict__ out) {
  __shared__ u32 As[2][BM][LDAU];
  __shared__ u32 Bs[2][BN][LDBU];
  const int wave = threadIdx.x >> 5, lane = threadIdx.x & 31;
  const int m0 = blockIdx.y * BM;
  const int n0 = blockIdx.x * BN;

  v8f acc[8] = {};
  gemm_core(Cx, Wo, m0, n0, GemmSlots{As, Bs}, acc);

#pragma unroll
  for (int nt = 0; nt < 8; ++nt) {
    const int n = n0 + nt * 16 + (lane & 15);
    const float bv = bo[n];
#pragma unroll
    for (int r = 0; r < 8; ++r) {
      const int m = m0 + wave * 16 + ((lane >> 4) << 3) + r;
      const float val = acc[nt][r] + bv;
      out[(size_t)m * DD + n] = is_in[m] ? val : X[(size_t)m * DD + n];
    }
  }
}

// ---------------------------------------------------------------------------
extern "C" void kernel_launch(void* const* d_in, const int* in_sizes, int n_in,
                              void* d_out, int out_size, void* d_ws, size_t ws_size,
                              hipStream_t stream) {
  const float* x      = (const float*)d_in[0];   // [4,4096,512]
  const float* pos    = (const float*)d_in[1];   // [4,4096,3]
  const float* in_w   = (const float*)d_in[2];   // [1536,512]
  const float* in_b   = (const float*)d_in[3];   // [1536]
  const float* out_w  = (const float*)d_in[4];   // [512,512]
  const float* out_b  = (const float*)d_in[5];   // [512]
  float* out = (float*)d_out;

  char* ws = (char*)d_ws;
  int*   nbr   = (int*)ws;                                       // 1 MB
  int*   is_in = (int*)(ws + (1 << 20));                         // 64 KB
  float* Q     = (float*)(ws + (1 << 20) + (1 << 16));           // 32 MB
  float* Kp    = Q  + (size_t)ROWS * DD;                         // 32 MB
  float* Vp    = Kp + (size_t)ROWS * DD;                         // 32 MB
  float* ctx   = Vp + (size_t)ROWS * DD;                         // 32 MB

  topk_kernel<<<ROWS / 8, 256, 0, stream>>>(pos, nbr, is_in);
  qkv_gemm_kernel<<<dim3(3 * DD / BN, ROWS / BM), 256, 0, stream>>>(x, in_w, in_b, Q, Kp, Vp);
  attn_kernel<<<ROWS / 8, 256, 0, stream>>>(Q, Kp, Vp, nbr, ctx);
  out_gemm_kernel<<<dim3(DD / BN, ROWS / BM), 256, 0, stream>>>(ctx, out_w, out_b, x, is_in, out);
  (void)in_sizes; (void)n_in; (void)out_size; (void)ws_size;
}
